// Task1_bf_final_CSW2D_38878043963667
// MI455X (gfx1250) — matrix operand order repaired and checked by reading, not compiled, not run on hardware
//
#include <hip/hip_runtime.h>

// ---------------- constants ----------------
#define FS_      20832000.0f
#define FDEMOD_  5208000.0f
#define CS_      1540.0f
#define FNUM_    1.0f
#define MIN_W_   0.001f
#define EPS_     1e-05f
#define TWO_PI_  6.283185307179586f

#define NA 16
#define NE 128
#define NS 2048
#define HH 128
#define WW 256
#define PP (HH*WW)          // 32768 pixels
#define EE 128              // elements (conv spatial dim)

typedef _Float16 v16h __attribute__((ext_vector_type(16)));
typedef _Float16 v8h  __attribute__((ext_vector_type(8)));
typedef float    v8f  __attribute__((ext_vector_type(8)));

// ---- monotonic uint encoding of signed float (for atomic max) ----
__device__ __forceinline__ unsigned encMono(float f) {
    unsigned u = __float_as_uint(f);
    return (u & 0x80000000u) ? ~u : (u | 0x80000000u);
}
__device__ __forceinline__ float decMono(unsigned k) {
    unsigned u = (k & 0x80000000u) ? (k ^ 0x80000000u) : ~k;
    return __uint_as_float(u);
}

// ---- gfx1250 async global->LDS copy (16 B per lane), tracked by ASYNCcnt ----
__device__ __forceinline__ void async_copy_b128(unsigned lds_byte_off, const void* gaddr) {
    asm volatile("global_load_async_to_lds_b128 %0, %1, off"
                 :: "v"(lds_byte_off), "v"(gaddr)
                 : "memory");
}
__device__ __forceinline__ void wait_async0() {
    asm volatile("s_wait_asynccnt 0" ::: "memory");
}

// ---------------- init: zero the accumulator block ----------------
__global__ void k_init(float* ws) {
    if (threadIdx.x < 128) ws[threadIdx.x] = 0.0f;
}

// ---------------- global max of i^2+q^2 ----------------
__global__ void k_max(const float* __restrict__ idata, const float* __restrict__ qdata,
                      unsigned* maxsq_bits, int n) {
    float m = 0.0f;
    for (int i = blockIdx.x * blockDim.x + threadIdx.x; i < n; i += gridDim.x * blockDim.x) {
        float a = idata[i], b = qdata[i];
        m = fmaxf(m, a * a + b * b);
    }
    __shared__ float red[256];
    red[threadIdx.x] = m;
    __syncthreads();
    for (int s = 128; s > 0; s >>= 1) {
        if (threadIdx.x < (unsigned)s) red[threadIdx.x] = fmaxf(red[threadIdx.x], red[threadIdx.x + s]);
        __syncthreads();
    }
    if (threadIdx.x == 0) atomicMax(maxsq_bits, __float_as_uint(red[0]));  // values >= 0: raw bits monotonic
}

// ---------------- beamform: delay, gather, phase-rotate, apodize ----------------
// block = 128 threads (one per element), grid = PP (one per pixel)
__global__ void k_bf(const float* __restrict__ idata, const float* __restrict__ qdata,
                     const float* __restrict__ angles, const float* __restrict__ ele_pos,
                     const float* __restrict__ time_zero, const float* __restrict__ grid3,
                     const unsigned* __restrict__ maxsq_bits,
                     _Float16* __restrict__ x0,            // [P][2][E]
                     float* __restrict__ sumIR, float* __restrict__ sumQR) {
    const int p = blockIdx.x;
    const int e = threadIdx.x;

    const float inv_nrm = 1.0f / sqrtf(__uint_as_float(*maxsq_bits));
    const float px = grid3[p * 3 + 0], py = grid3[p * 3 + 1], pz = grid3[p * 3 + 2];
    const float ang = angles[0];
    const float t0  = time_zero[0];
    const float ex = ele_pos[e * 3 + 0], ey = ele_pos[e * 3 + 1], ez = ele_pos[e * 3 + 2];

    const float txdel = (px * sinf(ang) + pz * cosf(ang) + t0 * CS_) * (FS_ / CS_);
    const float dx = px - ex, dy = py - ey, dz = pz - ez;
    const float rxdel = sqrtf(dx * dx + dy * dy + dz * dz) * (FS_ / CS_);
    const float delay = txdel + rxdel;
    const float d0f = floorf(delay);
    const float frac = delay - d0f;
    const int d0 = (int)d0f;

    const float* isig = idata + (size_t)e * NS;   // event 0
    const float* qsig = qdata + (size_t)e * NS;
    auto gath = [&](const float* s, int idx) -> float {
        int c = idx < 0 ? 0 : (idx > NS - 1 ? NS - 1 : idx);
        float v = s[c];
        return (idx >= 0 && idx < NS) ? v : 0.0f;
    };
    const float ifoc = (gath(isig, d0) * (1.0f - frac) + gath(isig, d0 + 1) * frac) * inv_nrm;
    const float qfoc = (gath(qsig, d0) * (1.0f - frac) + gath(qsig, d0 + 1) * frac) * inv_nrm;

    const float tshift = delay * (1.0f / FS_) - pz * (2.0f / CS_);
    const float theta = TWO_PI_ * FDEMOD_ * tshift;
    const float ct = cosf(theta), st = sinf(theta);
    float ir = ifoc * ct - qfoc * st;
    float qr = qfoc * ct + ifoc * st;

    // apodization
    const float e0x = ele_pos[0];
    const float eLx = ele_pos[(NE - 1) * 3];
    const float vx = px - ex, vz = pz - ez;
    const bool mrx = (fabsf(vz) > FNUM_ * fabsf(vx)) || (fabsf(vx) <= MIN_W_) ||
                     ((vx >= MIN_W_) && (px <= e0x)) || ((vx <= -MIN_W_) && (px >= eLx));
    const float xproj = px - pz * tanf(ang);
    const bool mtx = (xproj >= e0x * 1.2f) && (xproj <= eLx * 1.2f);
    const float ap = (mrx ? 1.0f : 0.0f) * (mtx ? 1.0f : 0.0f);
    ir *= ap; qr *= ap;

    x0[((size_t)p * 2 + 0) * EE + e] = (_Float16)ir;
    x0[((size_t)p * 2 + 1) * EE + e] = (_Float16)qr;

    __shared__ float rs[128], qs[128];
    rs[e] = ir; qs[e] = qr;
    __syncthreads();
    for (int s = 64; s > 0; s >>= 1) {
        if (e < s) { rs[e] += rs[e + s]; qs[e] += qs[e + s]; }
        __syncthreads();
    }
    if (e == 0) { sumIR[p] = rs[0]; sumQR[p] = qs[0]; }
}

// ---------------- WMMA conv along E (per pixel row) ----------------
// out[n, e] = sum_{c,t} w[n][c][t] * in[c][e + t - PAD]    (n = 0..7)
// One wave per row; 8 waves / 256 threads per block; GEMM M=128, N=16(8 real), K=CIN*KT padded.
// Input row staged via global_load_async_to_lds_b128, then BN+ReLU+halo transform into the
// im2col tile (with one extra zeroed row so out-of-range K reads need no predication).
template <int CIN, int KT, int PAD, bool APPLY_BN>
__global__ void __launch_bounds__(256)
k_conv(const _Float16* __restrict__ xin,   // [P][CIN][E] (pre-BN if APPLY_BN)
       const float* __restrict__ w,        // [8][CIN][KT]
       const float* __restrict__ bn_mu, const float* __restrict__ bn_istd,
       const float* __restrict__ bn_g,  const float* __restrict__ bn_b,
       _Float16* __restrict__ yout,        // [P][8][E] raw conv output
       float* __restrict__ statSum, float* __restrict__ statSumSq) {
    constexpr int LROW   = EE + 2 * PAD;          // padded row length (== EE + KT - 1)
    constexpr int KTOT   = CIN * KT;
    constexpr int KSTEPS = (KTOT + 31) / 32;
    constexpr int TSZ    = (CIN + 1) * LROW;      // + one zeroed row for OOB K reads

    const int wave = threadIdx.x >> 5;
    const int lane = threadIdx.x & 31;
    const int p = blockIdx.x * 8 + wave;

    __shared__ __align__(16) _Float16 stage[8][CIN * EE];   // raw input rows (async-filled)
    __shared__ _Float16 tile[8][TSZ];                        // zero-haloed im2col source
    __shared__ float ssum[8], ssq[8];
    if (threadIdx.x < 8) { ssum[threadIdx.x] = 0.0f; ssq[threadIdx.x] = 0.0f; }

    // ---- async stage: copy this wave's raw row (CIN*EE f16) into LDS ----
    {
        constexpr int rowBytes = CIN * EE * (int)sizeof(_Float16);   // 512 or 2048
        const char* gsrc = (const char*)(xin + (size_t)p * CIN * EE);
        const unsigned lbase = (unsigned)(unsigned long long)&stage[wave][0];
        #pragma unroll
        for (int off = 0; off < rowBytes; off += 32 * 16) {
            const int o = off + lane * 16;
            async_copy_b128(lbase + o, gsrc + o);
        }
        wait_async0();
    }

    // per-channel BN params of the *previous* layer (folded into the transform)
    float mu[CIN], istd[CIN], gg[CIN], bb[CIN];
    if constexpr (APPLY_BN) {
        #pragma unroll
        for (int c = 0; c < CIN; ++c) { mu[c] = bn_mu[c]; istd[c] = bn_istd[c]; gg[c] = bn_g[c]; bb[c] = bn_b[c]; }
    }

    // ---- transform: stage -> zero-haloed tile (+ zero row at c == CIN) ----
    for (int idx = lane; idx < TSZ; idx += 32) {
        const int c = idx / LROW;
        const int pos = idx - c * LROW;
        float v = 0.0f;
        if (c < CIN) {
            const int src = pos - PAD;
            if (src >= 0 && src < EE) {
                v = (float)stage[wave][c * EE + src];
                if constexpr (APPLY_BN) v = fmaxf((v - mu[c]) * istd[c] * gg[c] + bb[c], 0.0f);
            }
        }
        tile[wave][idx] = (_Float16)v;
    }

    // ---- B fragments (weights), B k order used here (same as A): col n = lane&15, khalf = lane>>4,
    //      VGPR i holds K = khalf*16 + 2i + {0,1} ----
    const int ncol  = lane & 15;
    const int khalf = lane >> 4;
    v16h bfrag[KSTEPS];
    #pragma unroll
    for (int s = 0; s < KSTEPS; ++s) {
        #pragma unroll
        for (int h = 0; h < 16; ++h) {
            const int i = h >> 1, j = h & 1;
            const int k = s * 32 + khalf * 16 + i * 2 + j;
            const int kk = (k < KTOT) ? k : 0;          // clamp: load unconditional
            const int c = kk / KT, t = kk - c * KT;
            const int nn = ncol & 7;
            float wv = w[((size_t)nn * CIN + c) * KT + t];
            wv = (ncol < 8 && k < KTOT) ? wv : 0.0f;    // select on value
            bfrag[s][h] = (_Float16)wv;
        }
    }
    __syncthreads();

    const int ahalf = khalf;               // A: lanes 0-15 and 16-31 both hold rows M=0..15
    float locSum = 0.0f, locSq = 0.0f;

    #pragma unroll
    for (int mt = 0; mt < 8; ++mt) {
        const int mrow = mt * 16 + (lane & 15);
        v8f acc = {};
        #pragma unroll
        for (int s = 0; s < KSTEPS; ++s) {
            // A k order used here (same as B): VGPR i, pair j -> Kloc = 16*ahalf + 2i + j
            v16h afrag;
            #pragma unroll
            for (int h = 0; h < 16; ++h) {
                const int i = h >> 1, j = h & 1;
                const int kloc = 16 * ahalf + 2 * i + j;
                const int k = s * 32 + kloc;
                const int c  = k / KT;
                const int a1 = c * LROW + mrow + (k - c * KT);
                const int a0 = CIN * LROW + mrow;            // zeroed row
                const int addr = (k < KTOT) ? a1 : a0;       // cndmask, load unconditional
                afrag[h] = tile[wave][addr];
            }
            acc = __builtin_amdgcn_wmma_f32_16x16x32_f16(false, afrag, false, bfrag[s],
                                                         (short)0, acc, false, false);
        }
        // D f32 layout: VGPR r -> M = (lane>>4)*8 + r (within tile), N = lane&15
        v8h dh;
        #pragma unroll
        for (int r = 0; r < 8; ++r) {
            const float d = acc[r];
            dh[r] = (_Float16)d;
            if (ncol < 8) { locSum += d; locSq += d * d; }
        }
        if (ncol < 8) {
            const int m0 = mt * 16 + (lane >> 4) * 8;
            *(v8h*)&yout[((size_t)p * 8 + ncol) * EE + m0] = dh;   // 16-B aligned b128 store
        }
    }
    if (ncol < 8) { atomicAdd(&ssum[ncol], locSum); atomicAdd(&ssq[ncol], locSq); }
    __syncthreads();
    if (threadIdx.x < 8) {
        atomicAdd(&statSum[threadIdx.x], ssum[threadIdx.x]);
        atomicAdd(&statSumSq[threadIdx.x], ssq[threadIdx.x]);
    }
}

// ---------------- finalize BN stats: sums -> mean / inv-std ----------------
__global__ void k_stats(const float* __restrict__ sum, const float* __restrict__ sumsq,
                        float* __restrict__ mu, float* __restrict__ istd,
                        int nch, float count) {
    const int c = threadIdx.x;
    if (c < nch) {
        const float m = sum[c] / count;
        const float v = sumsq[c] / count - m * m;
        mu[c] = m;
        istd[c] = rsqrtf(v + EPS_);
    }
}

// ---------------- conv4: 8->1 channels, 3 taps (scalar) ----------------
__global__ void k_conv4(const _Float16* __restrict__ y3,    // [P][8][E] pre-BN3
                        const float* __restrict__ w4,       // [1][8][1][3]
                        const float* __restrict__ mu3, const float* __restrict__ istd3,
                        const float* __restrict__ g3,  const float* __restrict__ b3,
                        float* __restrict__ w4raw,          // [P][E]
                        float* __restrict__ sum4, float* __restrict__ ssq4) {
    float locS = 0.0f, locQ = 0.0f;
    const int n = PP * EE;
    for (int idx = blockIdx.x * blockDim.x + threadIdx.x; idx < n; idx += gridDim.x * blockDim.x) {
        const int p = idx / EE, e = idx - p * EE;
        float acc = 0.0f;
        #pragma unroll
        for (int c = 0; c < 8; ++c) {
            const float m = mu3[c], is = istd3[c], g = g3[c], b = b3[c];
            #pragma unroll
            for (int t = 0; t < 3; ++t) {
                const int src = e + t - 1;
                if (src >= 0 && src < EE) {
                    float v = (float)y3[((size_t)p * 8 + c) * EE + src];
                    v = fmaxf((v - m) * is * g + b, 0.0f);
                    acc += w4[c * 3 + t] * v;
                }
            }
        }
        w4raw[idx] = acc;
        locS += acc; locQ += acc * acc;
    }
    __shared__ float rs[256], rq[256];
    rs[threadIdx.x] = locS; rq[threadIdx.x] = locQ;
    __syncthreads();
    for (int s = 128; s > 0; s >>= 1) {
        if (threadIdx.x < (unsigned)s) { rs[threadIdx.x] += rs[threadIdx.x + s]; rq[threadIdx.x] += rq[threadIdx.x + s]; }
        __syncthreads();
    }
    if (threadIdx.x == 0) { atomicAdd(sum4, rs[0]); atomicAdd(ssq4, rq[0]); }
}

// ---------------- final: BN4+ReLU, singleW, compound, dB, global max ----------------
__global__ void k_final(const float* __restrict__ w4raw,
                        const float* __restrict__ mu4, const float* __restrict__ istd4,
                        const float* __restrict__ g4,  const float* __restrict__ b4,
                        const float* __restrict__ sumIR, const float* __restrict__ sumQR,
                        float* __restrict__ out, unsigned* __restrict__ ymax_bits) {
    const int p = blockIdx.x;
    const int e = threadIdx.x;
    float v = w4raw[(size_t)p * EE + e];
    v = fmaxf((v - mu4[0]) * istd4[0] * g4[0] + b4[0], 0.0f);
    __shared__ float rs[128];
    rs[e] = v;
    __syncthreads();
    for (int s = 64; s > 0; s >>= 1) {
        if (e < s) rs[e] += rs[e + s];
        __syncthreads();
    }
    if (e == 0) {
        const float singleW = rs[0];
        const float ci = singleW * sumIR[p] * (1.0f / EE);
        const float cq = singleW * sumQR[p] * (1.0f / EE);
        const float mag = sqrtf(ci * ci + cq * cq);
        const float ydb = 20.0f * log10f(mag + 1e-20f);
        out[p] = ydb;
        atomicMax(ymax_bits, encMono(ydb));
    }
}

__global__ void k_sub(float* __restrict__ out, const unsigned* __restrict__ ymax_bits) {
    const float m = decMono(*ymax_bits);
    const int i = blockIdx.x * blockDim.x + threadIdx.x;
    if (i < PP) out[i] -= m;
}

// ---------------- host ----------------
extern "C" void kernel_launch(void* const* d_in, const int* in_sizes, int n_in,
                              void* d_out, int out_size, void* d_ws, size_t ws_size,
                              hipStream_t stream) {
    const float* idata    = (const float*)d_in[0];
    const float* qdata    = (const float*)d_in[1];
    const float* angles   = (const float*)d_in[2];
    const float* ele_pos  = (const float*)d_in[3];
    const float* tz       = (const float*)d_in[4];
    const float* grid3    = (const float*)d_in[5];
    const float* w1 = (const float*)d_in[6];
    const float* g1 = (const float*)d_in[8];  const float* be1 = (const float*)d_in[9];
    const float* w2 = (const float*)d_in[10];
    const float* g2 = (const float*)d_in[12]; const float* be2 = (const float*)d_in[13];
    const float* w3 = (const float*)d_in[14];
    const float* g3 = (const float*)d_in[16]; const float* be3 = (const float*)d_in[17];
    const float* w4 = (const float*)d_in[18];
    const float* g4 = (const float*)d_in[20]; const float* be4 = (const float*)d_in[21];
    // conv biases (d_in[7,11,15,19]) cancel exactly through batchnorm (mean absorbs them) -> skipped.

    float* ws = (float*)d_ws;
    unsigned* maxsq = (unsigned*)(ws + 0);
    unsigned* ymax  = (unsigned*)(ws + 1);
    float* sum1 = ws + 2,  *ssq1 = ws + 10;
    float* sum2 = ws + 18, *ssq2 = ws + 26;
    float* sum3 = ws + 34, *ssq3 = ws + 42;
    float* sum4 = ws + 50, *ssq4 = ws + 51;
    float* mu1 = ws + 58,  *istd1 = ws + 66;
    float* mu2 = ws + 74,  *istd2 = ws + 82;
    float* mu3 = ws + 90,  *istd3 = ws + 98;
    float* mu4 = ws + 106, *istd4 = ws + 107;

    char* base = (char*)d_ws + 512;
    const size_t X0B  = (size_t)PP * 2 * EE * sizeof(_Float16);   // 16.8 MB
    const size_t BUFB = (size_t)PP * 8 * EE * sizeof(_Float16);   // 67 MB
    _Float16* x0   = (_Float16*)base;
    float*    w4raw = (float*)base;                               // reuses x0 region (same size)
    _Float16* bufA = (_Float16*)(base + X0B);
    _Float16* bufB = (_Float16*)(base + X0B + BUFB);
    float* sumIR = (float*)(base + X0B + 2 * BUFB);
    float* sumQR = sumIR + PP;

    const float count = (float)((size_t)PP * EE);

    k_init<<<1, 256, 0, stream>>>(ws);
    k_max<<<1024, 256, 0, stream>>>(idata, qdata, maxsq, NA * NE * NS);
    k_bf<<<PP, 128, 0, stream>>>(idata, qdata, angles, ele_pos, tz, grid3, maxsq, x0, sumIR, sumQR);

    k_conv<2, 65, 32, false><<<PP / 8, 256, 0, stream>>>(x0, w1, nullptr, nullptr, nullptr, nullptr,
                                                         bufA, sum1, ssq1);
    k_stats<<<1, 32, 0, stream>>>(sum1, ssq1, mu1, istd1, 8, count);
    k_conv<8, 15, 7, true><<<PP / 8, 256, 0, stream>>>(bufA, w2, mu1, istd1, g1, be1,
                                                       bufB, sum2, ssq2);
    k_stats<<<1, 32, 0, stream>>>(sum2, ssq2, mu2, istd2, 8, count);
    k_conv<8, 15, 7, true><<<PP / 8, 256, 0, stream>>>(bufB, w3, mu2, istd2, g2, be2,
                                                       bufA, sum3, ssq3);
    k_stats<<<1, 32, 0, stream>>>(sum3, ssq3, mu3, istd3, 8, count);

    k_conv4<<<2048, 256, 0, stream>>>(bufA, w4, mu3, istd3, g3, be3, w4raw, sum4, ssq4);
    k_stats<<<1, 32, 0, stream>>>(sum4, ssq4, mu4, istd4, 1, count);

    k_final<<<PP, 128, 0, stream>>>(w4raw, mu4, istd4, g4, be4, sumIR, sumQR, (float*)d_out, ymax);
    k_sub<<<PP / 256, 256, 0, stream>>>((float*)d_out, ymax);
}
